// DpamNet_20409684590757
// MI455X (gfx1250) — matrix operand order repaired and checked by reading, not compiled, not run on hardware
//
#include <hip/hip_runtime.h>
#include <hip/hip_bf16.h>
#include <math.h>

// ---------------------------------------------------------------------------
// DpamNet forward for MI455X (gfx1250, wave32, WMMA).
// Heavy einsums run through v_wmma_f32_16x16x32_bf16 (bf16 in, f32 acc):
// bf16 does K=32 per WMMA vs K=4 for f32 -> ~8x matrix ceiling, and the
// per-batch 4096x4096 attention matrix (33.5 MB bf16) lives in the 192 MB L2
// across its three reuses, so the workload is matrix-op bound, not HBM bound.
// GEMM v4: 64x128 block tile, 4 waves x (16x128) strips -> 8 WMMAs per wave
// per K-step off one A fragment; double-buffered LDS B staging (one barrier
// per K-step, staging overlapped with the WMMA burst). K%32==0 via padding.
// ---------------------------------------------------------------------------

typedef __bf16 bhalf;
typedef __attribute__((ext_vector_type(8)))  __bf16 bf16x8;
typedef __attribute__((ext_vector_type(16))) __bf16 bf16x16;
typedef __attribute__((ext_vector_type(8)))  float  f32x8;

#define B_SZ 4

// ---------------------------------------------------------------------------
// Y(MxN, f32, ldy) = op(A) * op(B), bf16 operands, K % 32 == 0.
//   TA==0: A[m][k] = Ab[m*lda + k]    TA==1: A[m][k] = Ab[k*lda + m]
//   TB==0: B[k][n] = Bb[k*ldb + n]    TB==1: B[k][n] = Bb[n*ldb + k]
// Block tile 64(M) x 128(N): wave w owns rows [tM, tM+16) x all 128 cols
// (8 accumulators). M multiple of 64, N multiple of 128.
// Fragment layouts per CDNA5 ISA 7.12.2:
//   A 16x32 bf16: lane l -> row (l&15); K-base = (l<16 ? 0 : 8);
//     a[0..7] = K base+0..7, a[8..15] = K base+16..23    (two b128 runs)
//   B 32x16 bf16: lane l -> col (l&15); K-base = (l<16 ? 0 : 8);
//     b[0..7] = K base+0..7, b[8..15] = K base+16..23
//   C/D 16x16 f32: lane l -> col (l&15); rows (l<16 ? 0..7 : 8..15)
// ---------------------------------------------------------------------------
template<int TA, int TB>
__global__ __launch_bounds__(128)
void gemm_wmma(const bhalf* __restrict__ Ab, const bhalf* __restrict__ Bb,
               float* __restrict__ Yb, int K, int lda, int ldb, int ldy,
               long sA, long sB, long sY)
{
  const bhalf* A = Ab + (long)blockIdx.z * sA;
  const bhalf* B = Bb + (long)blockIdx.z * sB;
  float*       Y = Yb + (long)blockIdx.z * sY;

  const int lane = threadIdx.x;                  // 0..31
  const int tM   = (blockIdx.y * 4 + threadIdx.y) * 16;
  const int tN   = blockIdx.x * 128;
  const int half = lane >> 4;
  const int l15  = lane & 15;
  const int am   = tM + l15;                     // logical A row
  const int tid  = threadIdx.y * 32 + lane;      // 0..127

  constexpr int LDT = 40;                        // padded LDS stride: 16B-aligned
                                                 // b128 column reads, banks disjoint
  __shared__ bhalf Bs[(TB == 0) ? 2 * 128 * LDT : 1];

  f32x8 acc[8];
#pragma unroll
  for (int nt = 0; nt < 8; ++nt) acc[nt] = f32x8{0.f,0.f,0.f,0.f,0.f,0.f,0.f,0.f};

  const int kbA = half ? 8 : 0;
  const int kbB = half ? 8 : 0;

  // Stage a 32x128 B tile transposed into LDS buffer `buf` (TB==0 only).
  auto stage = [&](int ks, int buf) {
    bhalf* bs = &Bs[buf * 128 * LDT];
#pragma unroll
    for (int p = 0; p < 4; ++p) {
      const int run = tid + p * 128;             // 0..511
      const int r   = run >> 4;                  // K row 0..31
      const int cb  = (run & 15) * 8;            // col run base 0..120
      const bf16x8 v = *(const bf16x8*)(B + (long)(ks + r) * ldb + tN + cb);
#pragma unroll
      for (int j = 0; j < 8; ++j) bs[(cb + j) * LDT + r] = v[j];
    }
  };

  if constexpr (TB == 0) stage(0, 0);
  int cur = 0;

  for (int k0 = 0; k0 < K; k0 += 32) {
    // ---------------- A fragment (shared across the 8 N-tiles) -------------
    bf16x16 a;
    if constexpr (TA == 0) {
      const bhalf* ap = A + (long)am * lda + k0 + kbA;
      const bf16x8 lo = *(const bf16x8*)(ap);        // K base+0..7
      const bf16x8 hi = *(const bf16x8*)(ap + 16);   // K base+16..23
#pragma unroll
      for (int j = 0; j < 8; ++j) { a[j] = lo[j]; a[8 + j] = hi[j]; }
    } else {
      const int kb = k0 + kbA;
#pragma unroll
      for (int j = 0; j < 8; ++j) {                  // coalesced across lanes
        a[j]     = A[(long)(kb + j)      * lda + am];
        a[8 + j] = A[(long)(kb + 16 + j) * lda + am];
      }
    }

    if constexpr (TB == 0) {
      __syncthreads();                           // buf `cur` ready; prior reads done
      if (k0 + 32 < K) stage(k0 + 32, cur ^ 1);  // overlap staging with WMMAs below
    }

    // ---------------- 8 WMMAs off one A fragment ---------------------------
#pragma unroll
    for (int nt = 0; nt < 8; ++nt) {
      bf16x16 b;
      if constexpr (TB == 0) {
        const bhalf* bp = &Bs[cur * 128 * LDT + (nt * 16 + l15) * LDT + kbB];
        const bf16x8 lo = *(const bf16x8*)(bp);
        const bf16x8 hi = *(const bf16x8*)(bp + 16);
#pragma unroll
        for (int j = 0; j < 8; ++j) { b[j] = lo[j]; b[8 + j] = hi[j]; }
      } else {
        const bhalf* bp = B + (long)(tN + nt * 16 + l15) * ldb + k0 + kbB;
        const bf16x8 lo = *(const bf16x8*)(bp);
        const bf16x8 hi = *(const bf16x8*)(bp + 16);
#pragma unroll
        for (int j = 0; j < 8; ++j) { b[j] = lo[j]; b[8 + j] = hi[j]; }
      }
      acc[nt] = __builtin_amdgcn_wmma_f32_16x16x32_bf16(false, a, false, b,
                                                        (short)0, acc[nt], false, false);
    }
    cur ^= 1;
  }

  const int m0 = tM + (half ? 8 : 0);
#pragma unroll
  for (int nt = 0; nt < 8; ++nt) {
    const int bn = tN + nt * 16 + l15;
#pragma unroll
    for (int v = 0; v < 8; ++v)
      Y[(long)(m0 + v) * ldy + bn] = acc[nt][v];
  }
}

// --------------------------- elementwise / padding -------------------------
__global__ void cvt_f32_bf16(const float* __restrict__ x, bhalf* __restrict__ y, long n)
{
  long i = (long)blockIdx.x * blockDim.x + threadIdx.x;
  if (i < n) y[i] = (bhalf)x[i];
}

// (batches, rowsSrc, cols) f32 -> (batches, rowsDst, cols) bf16, zero-pad rows.
__global__ void pad_rows_bf16(const float* __restrict__ src, bhalf* __restrict__ dst,
                              int rowsSrc, int rowsDst, int cols, long n)
{
  long i = (long)blockIdx.x * blockDim.x + threadIdx.x;
  if (i >= n) return;
  const int  col = (int)(i % cols);
  const long t   = i / cols;
  const int  r   = (int)(t % rowsDst);
  const long b   = t / rowsDst;
  dst[i] = (r < rowsSrc) ? (bhalf)src[((long)b * rowsSrc + r) * cols + col] : (bhalf)0.f;
}

// (rows, colsSrc) f32 -> (rows, colsDst) bf16, zero-pad columns.
__global__ void pad_cols_bf16(const float* __restrict__ src, bhalf* __restrict__ dst,
                              int colsSrc, int colsDst, long n)
{
  long i = (long)blockIdx.x * blockDim.x + threadIdx.x;
  if (i >= n) return;
  const long r = i / colsDst;
  const int  c = (int)(i % colsDst);
  dst[i] = (c < colsSrc) ? (bhalf)src[r * colsSrc + c] : (bhalf)0.f;
}

// Row softmax on f32 Gram -> bf16 attention row (softmax over last axis).
__global__ void softmax_row_bf16(const float* __restrict__ G, bhalf* __restrict__ A, int Nc)
{
  const int row = blockIdx.x, t = threadIdx.x;
  const float* g = G + (long)row * Nc;
  bhalf*       a = A + (long)row * Nc;
  __shared__ float red[256];
  float mx = -3.4e38f;
  for (int j = t; j < Nc; j += 256) mx = fmaxf(mx, g[j]);
  red[t] = mx; __syncthreads();
  for (int s = 128; s; s >>= 1) { if (t < s) red[t] = fmaxf(red[t], red[t + s]); __syncthreads(); }
  mx = red[0]; __syncthreads();
  float sum = 0.f;
  for (int j = t; j < Nc; j += 256) sum += __expf(g[j] - mx);
  red[t] = sum; __syncthreads();
  for (int s = 128; s; s >>= 1) { if (t < s) red[t] += red[t + s]; __syncthreads(); }
  const float inv = 1.f / red[0];
  for (int j = t; j < Nc; j += 256) a[j] = (bhalf)(__expf(g[j] - mx) * inv);
}

// Channel softmax (axis=channel) on bf16 (Co x Np), one thread per point.
__global__ void softmax_chan_bf16(const bhalf* __restrict__ X, bhalf* __restrict__ S,
                                  int Co, int Np)
{
  const int n = blockIdx.x * blockDim.x + threadIdx.x;
  const int b = blockIdx.y;
  if (n >= Np) return;
  const bhalf* x = X + (long)b * Co * Np + n;
  bhalf*       s = S + (long)b * Co * Np + n;
  float mx = -3.4e38f;
  for (int c = 0; c < Co; ++c) mx = fmaxf(mx, (float)x[(long)c * Np]);
  float sum = 0.f;
  for (int c = 0; c < Co; ++c) sum += __expf((float)x[(long)c * Np] - mx);
  const float inv = 1.f / sum;
  for (int c = 0; c < Co; ++c) s[(long)c * Np] = (bhalf)(__expf((float)x[(long)c * Np] - mx) * inv);
}

// Per-channel mean / 1/sqrt(var+eps) over (batch, points). Biased var (jnp.var).
__global__ void bn_stats(const float* __restrict__ Y, float* __restrict__ mean,
                         float* __restrict__ istd, int O, int Np)
{
  const int o = blockIdx.x, t = threadIdx.x;
  __shared__ float s1[256], s2[256];
  float a = 0.f, b = 0.f;
  for (int bb = 0; bb < B_SZ; ++bb) {
    const float* p = Y + ((long)bb * O + o) * Np;
    for (int n = t; n < Np; n += 256) { float v = p[n]; a += v; b += v * v; }
  }
  s1[t] = a; s2[t] = b; __syncthreads();
  for (int s = 128; s; s >>= 1) {
    if (t < s) { s1[t] += s1[t + s]; s2[t] += s2[t + s]; }
    __syncthreads();
  }
  if (!t) {
    const float cnt = (float)B_SZ * (float)Np;
    const float m = s1[0] / cnt;
    const float v = s2[0] / cnt - m * m;
    mean[o] = m; istd[o] = rsqrtf(v + 1e-5f);
  }
}

__global__ void bn_apply_relu(const float* __restrict__ Y, const float* __restrict__ mean,
                              const float* __restrict__ istd, const float* __restrict__ g,
                              const float* __restrict__ bt, bhalf* __restrict__ X,
                              int O, int Np)
{
  const long tot = (long)O * Np;
  long i = (long)blockIdx.x * blockDim.x + threadIdx.x;
  if (i >= tot) return;
  const int o = (int)(i / Np);
  const long bi = (long)blockIdx.y * tot + i;
  const float v = (Y[bi] - mean[o]) * istd[o] * g[o] + bt[o];
  X[bi] = (bhalf)fmaxf(v, 0.f);
}

// Global max over points, write into concatenated descriptor slot.
__global__ void pool_max_bf16(const bhalf* __restrict__ X, float* __restrict__ H,
                              int C, int Np, int off, int hld)
{
  const int c = blockIdx.x, b = blockIdx.y, t = threadIdx.x;
  const bhalf* x = X + ((long)b * C + c) * Np;
  __shared__ float red[256];
  float m = -3.4e38f;
  for (int n = t; n < Np; n += 256) m = fmaxf(m, (float)x[n]);
  red[t] = m; __syncthreads();
  for (int s = 128; s; s >>= 1) { if (t < s) red[t] = fmaxf(red[t], red[t + s]); __syncthreads(); }
  if (!t) H[(long)b * hld + off + c] = red[0];
}

// l = || I - s s^T ||_F  given G = s s^T (Co x Co, f32) per batch.
__global__ void mnorm_l(const float* __restrict__ G, float* __restrict__ lpart, int Co)
{
  const int b = blockIdx.x, t = threadIdx.x;
  const float* g = G + (long)b * Co * Co;
  __shared__ float red[256];
  float acc = 0.f;
  const long tot = (long)Co * Co;
  for (long i = t; i < tot; i += 256) {
    const int c = (int)(i / Co), d = (int)(i % Co);
    const float m = ((c == d) ? 1.f : 0.f) - g[i];
    acc += m * m;
  }
  red[t] = acc; __syncthreads();
  for (int s = 128; s; s >>= 1) { if (t < s) red[t] += red[t + s]; __syncthreads(); }
  if (!t) lpart[b] = sqrtf(red[0]);
}

// Small dense FC head (f32): Out[b][o] = sum_i H[b][i] * W[o][i]
__global__ void fc_dense(const float* __restrict__ H, const float* __restrict__ W,
                         float* __restrict__ Out, int I, int O)
{
  const int o = blockIdx.x * blockDim.x + threadIdx.x;
  const int b = blockIdx.y;
  if (o >= O) return;
  const float* h = H + (long)b * I;
  const float* w = W + (long)o * I;
  float s = 0.f;
  for (int i = 0; i < I; ++i) s += h[i] * w[i];
  Out[(long)b * O + o] = s;
}

// BatchNorm over the batch axis only (head), in place + ReLU.
__global__ void bn_head_relu(float* __restrict__ H, const float* __restrict__ g,
                             const float* __restrict__ bt, int O)
{
  const int o = blockIdx.x * blockDim.x + threadIdx.x;
  if (o >= O) return;
  float s = 0.f, ss = 0.f;
  for (int b = 0; b < B_SZ; ++b) { const float v = H[(long)b * O + o]; s += v; ss += v * v; }
  const float m = s / B_SZ, var = ss / B_SZ - m * m, is = rsqrtf(var + 1e-5f);
  for (int b = 0; b < B_SZ; ++b) {
    const float v = (H[(long)b * O + o] - m) * is * g[o] + bt[o];
    H[(long)b * O + o] = fmaxf(v, 0.f);
  }
}

__global__ void l_final(const float* __restrict__ lp, float* __restrict__ out)
{
  const int b = threadIdx.x;
  if (b < B_SZ) out[b] = lp[b] + lp[4 + b] + lp[8 + b];
}

// ---------------------------------------------------------------------------
static inline unsigned idivu(long a, long b) { return (unsigned)((a + b - 1) / b); }

extern "C" void kernel_launch(void* const* d_in, const int* in_sizes, int n_in,
                              void* d_out, int out_size, void* d_ws, size_t ws_size,
                              hipStream_t stream)
{
  (void)in_sizes; (void)n_in; (void)out_size; (void)ws_size;
  auto IN = [&](int i) { return (const float*)d_in[i]; };

  // ---- workspace carve (bump allocator, 256B aligned), ~320 MB peak ----
  char* base = (char*)d_ws;
  size_t off = 0;
  auto carve = [&](size_t bytes) -> void* {
    void* p = base + off;
    off = (off + bytes + 255) & ~(size_t)255;
    return p;
  };
  float* Yf   = (float*)carve((size_t)4 * 512 * 4096 * 4);   // GEMM f32 output
  bhalf* Tbf  = (bhalf*)carve((size_t)4 * 512 * 4096 * 2);   // xh@A result, bf16
  bhalf* Wbf  = (bhalf*)carve((size_t)1024 * 256 * 2);       // weight bf16 scratch
  bhalf* Xbf  = (bhalf*)carve((size_t)4 * 128 * 4096 * 2);   // stage features x
  bhalf* Hb0  = (bhalf*)carve((size_t)4 * 512 * 4096 * 2);   // dpam xh ping
  bhalf* Hb1  = (bhalf*)carve((size_t)4 * 512 * 4096 * 2);   // dpam xh pong
  bhalf* Sbf  = (bhalf*)carve((size_t)4 * 512 * 4096 * 2);   // s (channel softmax)
  float* Gram = (float*)carve((size_t)4096 * 4096 * 4);      // per-batch Gram (f32)
  bhalf* Att  = (bhalf*)carve((size_t)4 * 4096 * 4096 * 2);  // attention, all batches
  float* Gf   = (float*)carve((size_t)4 * 512 * 512 * 4);    // s s^T
  float* Mean = (float*)carve(1024 * 4);
  float* Istd = (float*)carve(1024 * 4);
  float* Hcat = (float*)carve(4 * 1536 * 4);
  float* H1   = (float*)carve(4 * 512 * 4);
  float* H2   = (float*)carve(4 * 256 * 4);
  float* Lp   = (float*)carve(16 * 4);

  auto gemm = [&](const bhalf* A, const bhalf* Bm, float* Y, int M, int N, int K,
                  int lda, int ldb, int ldy, int tA, int tB,
                  long sA, long sB, long sY, int nb) {
    dim3 g((unsigned)(N / 128), (unsigned)(M / 64), (unsigned)nb), blk(32, 4);
    if (tA == 0 && tB == 0)
      gemm_wmma<0, 0><<<g, blk, 0, stream>>>(A, Bm, Y, K, lda, ldb, ldy, sA, sB, sY);
    else if (tA == 1)
      gemm_wmma<1, 0><<<g, blk, 0, stream>>>(A, Bm, Y, K, lda, ldb, ldy, sA, sB, sY);
    else
      gemm_wmma<0, 1><<<g, blk, 0, stream>>>(A, Bm, Y, K, lda, ldb, ldy, sA, sB, sY);
  };
  auto cvt = [&](const float* x, bhalf* y, long n) {
    cvt_f32_bf16<<<idivu(n, 256), 256, 0, stream>>>(x, y, n);
  };
  // conv1x1 (W: OxC, C % 32 == 0) + BN(over batch&points) + ReLU, bf16 out.
  auto conv_bn = [&](const float* W, const float* g, const float* bt,
                     int O, int C, int Np, const bhalf* Xin, bhalf* Xout) {
    cvt(W, Wbf, (long)O * C);
    gemm(Wbf, Xin, Yf, O, Np, C, C, Np, Np, 0, 0, 0, (long)C * Np, (long)O * Np, B_SZ);
    bn_stats<<<O, 256, 0, stream>>>(Yf, Mean, Istd, O, Np);
    const long tot = (long)O * Np;
    bn_apply_relu<<<dim3(idivu(tot, 256), B_SZ), 256, 0, stream>>>(Yf, Mean, Istd, g, bt, Xout, O, Np);
  };
  // Full dpam block. x = Xbf (B,Cin,Np). Writes s.x back to Xbf as (B,Cin,Co).
  auto dpam = [&](const float* const* w, const float* const* gg, const float* const* bb,
                  const int* Od, int Cin, int Np, float* lpart) {
    // A = softmax(x^T x) per batch; Gram in f32 for softmax accuracy, A stored bf16.
    for (int b = 0; b < B_SZ; ++b) {
      gemm(Xbf + (long)b * Cin * Np, Xbf + (long)b * Cin * Np, Gram,
           Np, Np, Cin, Np, Np, Np, 1, 0, 0, 0, 0, 1);
      softmax_row_bf16<<<(unsigned)Np, 256, 0, stream>>>(Gram, Att + (long)b * Np * Np, Np);
    }
    const bhalf* cur = Xbf;
    int Cc = Cin;
    for (int i = 0; i < 5; ++i) {
      const bhalf* gin = cur;
      if (i < 3) {  // xh = xh @ A   (contract over A's first index)
        gemm(cur, Att, Yf, Cc, Np, Np, Np, Np, Np, 0, 0,
             (long)Cc * Np, (long)Np * Np, (long)Cc * Np, B_SZ);
        cvt(Yf, Tbf, (long)B_SZ * Cc * Np);
        gin = Tbf;
      }
      bhalf* dst = (i & 1) ? Hb1 : Hb0;
      cvt(w[i], Wbf, (long)Od[i] * Cc);
      gemm(Wbf, gin, Yf, Od[i], Np, Cc, Cc, Np, Np, 0, 0,
           0, (long)Cc * Np, (long)Od[i] * Np, B_SZ);
      bn_stats<<<(unsigned)Od[i], 256, 0, stream>>>(Yf, Mean, Istd, Od[i], Np);
      const long tot = (long)Od[i] * Np;
      bn_apply_relu<<<dim3(idivu(tot, 256), B_SZ), 256, 0, stream>>>(
          Yf, Mean, Istd, gg[i], bb[i], dst, Od[i], Np);
      cur = dst; Cc = Od[i];
    }
    const int Co = Cc;
    softmax_chan_bf16<<<dim3(idivu(Np, 256), B_SZ), 256, 0, stream>>>(cur, Sbf, Co, Np);
    // l = ||I - s s^T||_F
    gemm(Sbf, Sbf, Gf, Co, Co, Np, Np, Np, Co, 0, 1,
         (long)Co * Np, (long)Co * Np, (long)Co * Co, B_SZ);
    mnorm_l<<<B_SZ, 256, 0, stream>>>(Gf, lpart, Co);
    // x_new[b][c][o] = sum_n s[o][n] x[c][n]
    gemm(Xbf, Sbf, Yf, Cin, Co, Np, Np, Np, Co, 0, 1,
         (long)Cin * Np, (long)Co * Np, (long)Cin * Co, B_SZ);
    cvt(Yf, Xbf, (long)B_SZ * Cin * Co);
  };

  // ------------------------------ pipeline ---------------------------------
  // mlp1 layer 0 has K=3: zero-pad x channels (3->32) and W columns (3->32)
  // so every GEMM in the net runs the guard-free K%32==0 path.
  pad_rows_bf16<<<idivu((long)B_SZ * 32 * 4096, 256), 256, 0, stream>>>(
      IN(0), Hb0, 3, 32, 4096, (long)B_SZ * 32 * 4096);
  pad_cols_bf16<<<idivu(64 * 32, 256), 256, 0, stream>>>(IN(1), Wbf, 3, 32, (long)64 * 32);
  gemm(Wbf, Hb0, Yf, 64, 4096, 32, 32, 4096, 4096, 0, 0, 0, (long)32 * 4096, (long)64 * 4096, B_SZ);
  bn_stats<<<64, 256, 0, stream>>>(Yf, Mean, Istd, 64, 4096);
  bn_apply_relu<<<dim3(idivu((long)64 * 4096, 256), B_SZ), 256, 0, stream>>>(
      Yf, Mean, Istd, IN(2), IN(3), Xbf, 64, 4096);

  // mlp1 layer 1: 64->128 at N=4096
  conv_bn(IN(4), IN(5), IN(6), 128, 64, 4096, Xbf, Xbf);
  pool_max_bf16<<<dim3(128, B_SZ), 256, 0, stream>>>(Xbf, Hcat, 128, 4096, 0, 1536);

  // dpam1 (Cin=128, Np=4096, Co=512)
  {
    const float* w[5] = {IN(16), IN(19), IN(22), IN(25), IN(28)};
    const float* g[5] = {IN(17), IN(20), IN(23), IN(26), IN(29)};
    const float* b[5] = {IN(18), IN(21), IN(24), IN(27), IN(30)};
    const int   O[5] = {128, 256, 512, 128, 512};
    dpam(w, g, b, O, 128, 4096, Lp + 0);
  }

  // mlp2: 128->128 at N=512
  conv_bn(IN(7), IN(8), IN(9), 128, 128, 512, Xbf, Xbf);
  pool_max_bf16<<<dim3(128, B_SZ), 256, 0, stream>>>(Xbf, Hcat, 128, 512, 128, 1536);

  // dpam2 (Cin=128, Np=512, Co=256)
  {
    const float* w[5] = {IN(31), IN(34), IN(37), IN(40), IN(43)};
    const float* g[5] = {IN(32), IN(35), IN(38), IN(41), IN(44)};
    const float* b[5] = {IN(33), IN(36), IN(39), IN(42), IN(45)};
    const int   O[5] = {128, 256, 512, 128, 256};
    dpam(w, g, b, O, 128, 512, Lp + 4);
  }

  // mlp3: 128->256 at N=256
  conv_bn(IN(10), IN(11), IN(12), 256, 128, 256, Xbf, Xbf);
  pool_max_bf16<<<dim3(256, B_SZ), 256, 0, stream>>>(Xbf, Hcat, 256, 256, 256, 1536);

  // dpam3 (Cin=256, Np=256, Co=128)
  {
    const float* w[5] = {IN(46), IN(49), IN(52), IN(55), IN(58)};
    const float* g[5] = {IN(47), IN(50), IN(53), IN(56), IN(59)};
    const float* b[5] = {IN(48), IN(51), IN(54), IN(57), IN(60)};
    const int   O[5] = {128, 256, 512, 128, 128};
    dpam(w, g, b, O, 256, 256, Lp + 8);
  }

  // mlp4: 256->1024 at N=128
  conv_bn(IN(13), IN(14), IN(15), 1024, 256, 128, Xbf, Xbf);
  pool_max_bf16<<<dim3(1024, B_SZ), 256, 0, stream>>>(Xbf, Hcat, 1024, 128, 512, 1536);

  // head: fcdown(512x1536)+BN+ReLU, fcfeat(256x512)+BN+ReLU, fc(40x256)
  fc_dense<<<dim3(idivu(512, 128), B_SZ), 128, 0, stream>>>(Hcat, IN(61), H1, 1536, 512);
  bn_head_relu<<<idivu(512, 256), 256, 0, stream>>>(H1, IN(64), IN(65), 512);
  fc_dense<<<dim3(idivu(256, 128), B_SZ), 128, 0, stream>>>(H1, IN(62), H2, 512, 256);
  bn_head_relu<<<1, 256, 0, stream>>>(H2, IN(66), IN(67), 256);
  fc_dense<<<dim3(1, B_SZ), 128, 0, stream>>>(H2, IN(63), (float*)d_out, 256, 40);
  l_final<<<1, 32, 0, stream>>>(Lp, (float*)d_out + 160);
}
